// Attention_ViT_8916352106583
// MI455X (gfx1250) — matrix operand order repaired and checked by reading, not compiled, not run on hardware
//
#include <hip/hip_runtime.h>
#include <hip/hip_bf16.h>

typedef __bf16 bf16;
typedef __attribute__((ext_vector_type(16))) __bf16 v16bf;
typedef __attribute__((ext_vector_type(8)))  float  v8f;

union Frag16 { v16bf v; uint4 q[2]; };

// A-fragment (16x32, contraction along k): row m = lane&15, halves split at h = lane>>4.
// Per ISA layout: element e -> k = 8h+e (e<8), k = 16+8h+(e-8) (e>=8). Two contiguous 16B runs.
__device__ __forceinline__ v16bf load_frag_a(const bf16* base, int ld, int lane) {
    const int m = lane & 15, hh = lane >> 4;
    Frag16 f;
    f.q[0] = *(const uint4*)(base + m * ld + 8 * hh);
    f.q[1] = *(const uint4*)(base + m * ld + 16 + 8 * hh);
    return f.v;
}

// B-fragment (32x16): base points at [n][k] layout, row stride ld.
// Per ISA layout: element e -> k = 8h+e (e<8), k = 16+8h+(e-8) (e>=8), n = lane&15 -> two contiguous 16B runs.
__device__ __forceinline__ v16bf load_frag_b(const bf16* base, int ld, int lane) {
    const int n = lane & 15, hh = lane >> 4;
    const bf16* p = base + n * ld + 8 * hh;
    Frag16 f;
    f.q[0] = *(const uint4*)(p);
    f.q[1] = *(const uint4*)(p + 16);
    return f.v;
}

#define WMMA_BF16(a, b, c) \
    __builtin_amdgcn_wmma_f32_16x16x32_bf16(false, (a), false, (b), (short)0, (c), false, false)

// ---------------------------------------------------------------------------
// Kernel 1: qkv = x @ w_qkv ; scatter to Q (scaled by dk^-0.5), K, V^T in bf16.
// Block tile 128x128, 8 waves, wave tile 32x64. K-step 32, double-buffered LDS.
// ---------------------------------------------------------------------------
__global__ __launch_bounds__(256)
void qkv_gemm_kernel(const float* __restrict__ x, const float* __restrict__ w,
                     bf16* __restrict__ Qw, bf16* __restrict__ Kw, bf16* __restrict__ Vt) {
    __shared__ bf16 As[2][128][40];   // [m][k] bf16, +8 pad
    __shared__ bf16 Bs[2][128][40];   // [n][k] bf16 (transposed in LDS)

    const int tid  = threadIdx.x;
    const int lane = tid & 31;
    const int wave = tid >> 5;
    const int mw = wave >> 1, nw = wave & 1;
    const int mbase = blockIdx.x * 128;
    const int nbase = blockIdx.y * 128;

    v8f acc[2][4] = {};

    auto load_tile = [&](int kt, int buf) {
        const int k0 = kt * 32;
#pragma unroll
        for (int j = 0; j < 4; ++j) {
            int i = tid + 256 * j;                 // 1024 float4 for 128x32 A tile
            int row = i >> 3, kq = i & 7;
            float4 f = *(const float4*)(x + (size_t)(mbase + row) * 1024 + k0 + kq * 4);
            union { bf16 h[4]; uint2 u; } c;
            c.h[0] = (bf16)f.x; c.h[1] = (bf16)f.y; c.h[2] = (bf16)f.z; c.h[3] = (bf16)f.w;
            *(uint2*)&As[buf][row][kq * 4] = c.u;
        }
#pragma unroll
        for (int j = 0; j < 4; ++j) {
            int i = tid + 256 * j;                 // 32(k) x 128(n) B tile, store transposed
            int kr = i >> 5, nq = i & 31;
            float4 f = *(const float4*)(w + (size_t)(k0 + kr) * 3072 + nbase + nq * 4);
            Bs[buf][nq * 4 + 0][kr] = (bf16)f.x;
            Bs[buf][nq * 4 + 1][kr] = (bf16)f.y;
            Bs[buf][nq * 4 + 2][kr] = (bf16)f.z;
            Bs[buf][nq * 4 + 3][kr] = (bf16)f.w;
        }
    };

    auto prefetch_tile = [&](int kt) {             // pull kt's tile toward L2 early
        const int k0 = kt * 32;
        {
            int row = tid >> 1, kq = (tid & 1) * 16;
            __builtin_prefetch(x + (size_t)(mbase + row) * 1024 + k0 + kq, 0, 0);
        }
        {
            int kr = tid >> 3, nq = (tid & 7) * 16;
            __builtin_prefetch(w + (size_t)(k0 + kr) * 3072 + nbase + nq, 0, 0);
        }
    };

    prefetch_tile(1);
    load_tile(0, 0);
    __syncthreads();
    int buf = 0;
    for (int kt = 0; kt < 32; ++kt) {
        if (kt + 2 < 32) prefetch_tile(kt + 2);
        if (kt + 1 < 32) load_tile(kt + 1, buf ^ 1);
        // Preload all fragments, then run the WMMA chain (staggered s_wait_dscnt).
        v16bf a0 = load_frag_a(&As[buf][mw * 32][0], 40, lane);
        v16bf a1 = load_frag_a(&As[buf][mw * 32 + 16][0], 40, lane);
        v16bf b0 = load_frag_b(&Bs[buf][nw * 64 + 0][0], 40, lane);
        v16bf b1 = load_frag_b(&Bs[buf][nw * 64 + 16][0], 40, lane);
        v16bf b2 = load_frag_b(&Bs[buf][nw * 64 + 32][0], 40, lane);
        v16bf b3 = load_frag_b(&Bs[buf][nw * 64 + 48][0], 40, lane);
        acc[0][0] = WMMA_BF16(a0, b0, acc[0][0]);
        acc[1][0] = WMMA_BF16(a1, b0, acc[1][0]);
        acc[0][1] = WMMA_BF16(a0, b1, acc[0][1]);
        acc[1][1] = WMMA_BF16(a1, b1, acc[1][1]);
        acc[0][2] = WMMA_BF16(a0, b2, acc[0][2]);
        acc[1][2] = WMMA_BF16(a1, b2, acc[1][2]);
        acc[0][3] = WMMA_BF16(a0, b3, acc[0][3]);
        acc[1][3] = WMMA_BF16(a1, b3, acc[1][3]);
        buf ^= 1;
        __syncthreads();
    }

    // Epilogue: C element (M = r + 8h, N = lane&15). Scatter to Q/K/V^T bf16.
    const int hh = lane >> 4, nL = lane & 15;
#pragma unroll
    for (int fn = 0; fn < 4; ++fn) {
        const int col = nbase + nw * 64 + fn * 16 + nL;
        const int t = col >> 10, rem = col & 1023;
        const int head = rem >> 6, d = rem & 63;
#pragma unroll
        for (int fm = 0; fm < 2; ++fm) {
#pragma unroll
            for (int r = 0; r < 8; ++r) {
                const int row = mbase + mw * 32 + fm * 16 + r + 8 * hh;
                const int b = row >> 10, n = row & 1023;
                const size_t bh = (size_t)(b * 16 + head);
                float v = acc[fm][fn][r];
                if (t == 0)      Qw[(bh * 1024 + n) * 64 + d] = (bf16)(v * 0.125f); // dk^-0.5
                else if (t == 1) Kw[(bh * 1024 + n) * 64 + d] = (bf16)v;
                else             Vt[(bh * 64 + d) * 1024 + n] = (bf16)v;            // V transposed
            }
        }
    }
}

// ---------------------------------------------------------------------------
// Kernel 2: flash attention. One wave per (b,h, 16-query block). 32 keys/iter.
// ---------------------------------------------------------------------------
__global__ __launch_bounds__(32)
void attn_kernel(const bf16* __restrict__ Q, const bf16* __restrict__ K,
                 const bf16* __restrict__ Vt, bf16* __restrict__ AO) {
    __shared__ bf16 Pt[16][40];       // P tile for C-layout -> A-layout conversion
    const int lane = threadIdx.x & 31;
    const int hh = lane >> 4, nL = lane & 15;
    const int wid = blockIdx.x;
    const int qb = wid & 63;          // query block (16 rows)
    const int bh = wid >> 6;          // b*16 + h
    const bf16* Qp = Q  + ((size_t)bh * 1024 + qb * 16) * 64;
    const bf16* Kp = K  + (size_t)bh * 1024 * 64;
    const bf16* Vp = Vt + (size_t)bh * 64 * 1024;

    const v16bf q0 = load_frag_a(Qp, 64, lane);       // dk 0..31
    const v16bf q1 = load_frag_a(Qp + 32, 64, lane);  // dk 32..63

    v8f o[4] = {};
    float mrow[8], lrow[8];
#pragma unroll
    for (int r = 0; r < 8; ++r) { mrow[r] = -1e30f; lrow[r] = 0.f; }

    for (int kb = 0; kb < 1024; kb += 32) {
        if (kb + 32 < 1024) {         // prefetch next K/V blocks toward L2
            __builtin_prefetch(Kp + (size_t)(kb + 32 + lane) * 64, 0, 0);
            __builtin_prefetch(Vp + (size_t)(lane * 2) * 1024 + kb + 32, 0, 0);
        }
        v8f s0 = {}, s1 = {};
        {   // S = Q @ K^T : K rows are contiguous-dk -> direct B fragments from global
            v16bf b00 = load_frag_b(Kp + (size_t)kb * 64, 64, lane);
            v16bf b01 = load_frag_b(Kp + (size_t)kb * 64 + 32, 64, lane);
            v16bf b10 = load_frag_b(Kp + (size_t)(kb + 16) * 64, 64, lane);
            v16bf b11 = load_frag_b(Kp + (size_t)(kb + 16) * 64 + 32, 64, lane);
            s0 = WMMA_BF16(q0, b00, s0);
            s0 = WMMA_BF16(q1, b01, s0);
            s1 = WMMA_BF16(q0, b10, s1);
            s1 = WMMA_BF16(q1, b11, s1);
        }
        float alpha[8];
#pragma unroll
        for (int r = 0; r < 8; ++r) {
            // row (r+8h) stats live in 16-lane groups -> width-16 xor reductions
            float mx = fmaxf(s0[r], s1[r]);
            mx = fmaxf(mx, __shfl_xor(mx, 1, 16));
            mx = fmaxf(mx, __shfl_xor(mx, 2, 16));
            mx = fmaxf(mx, __shfl_xor(mx, 4, 16));
            mx = fmaxf(mx, __shfl_xor(mx, 8, 16));
            const float mnew = fmaxf(mrow[r], mx);
            alpha[r] = __expf(mrow[r] - mnew);
            mrow[r] = mnew;
            const float p0 = __expf(s0[r] - mnew);
            const float p1 = __expf(s1[r] - mnew);
            float sum = p0 + p1;
            sum += __shfl_xor(sum, 1, 16);
            sum += __shfl_xor(sum, 2, 16);
            sum += __shfl_xor(sum, 4, 16);
            sum += __shfl_xor(sum, 8, 16);
            lrow[r] = lrow[r] * alpha[r] + sum;
            Pt[r + 8 * hh][nL]      = (bf16)p0;   // keys kb..kb+15
            Pt[r + 8 * hh][nL + 16] = (bf16)p1;   // keys kb+16..kb+31
        }
        asm volatile("s_wait_dscnt 0" ::: "memory");  // single wave: in-order DS turnaround
        const v16bf pf = load_frag_a(&Pt[0][0], 40, lane);
#pragma unroll
        for (int f = 0; f < 4; ++f) {
#pragma unroll
            for (int r = 0; r < 8; ++r) o[f][r] *= alpha[r];
            // V^T layout [d][n]: contiguous keys per lane -> direct B fragment
            v16bf vf = load_frag_b(Vp + (size_t)(f * 16) * 1024 + kb, 1024, lane);
            o[f] = WMMA_BF16(pf, vf, o[f]);
        }
    }

    const int b = bh >> 4, head = bh & 15;
#pragma unroll
    for (int f = 0; f < 4; ++f) {
        const int col = head * 64 + f * 16 + nL;
#pragma unroll
        for (int r = 0; r < 8; ++r) {
            const int row = qb * 16 + r + 8 * hh;
            AO[((size_t)b * 1024 + row) * 1024 + col] = (bf16)(o[f][r] / lrow[r]);
        }
    }
}

// ---------------------------------------------------------------------------
// Kernel 3: out = AO @ w_out + b_out (fp32 output). Same tiling as kernel 1.
// ---------------------------------------------------------------------------
__global__ __launch_bounds__(256)
void out_gemm_kernel(const bf16* __restrict__ A, const float* __restrict__ w,
                     const float* __restrict__ bias, float* __restrict__ out) {
    __shared__ bf16 As[2][128][40];
    __shared__ bf16 Bs[2][128][40];

    const int tid  = threadIdx.x;
    const int lane = tid & 31;
    const int wave = tid >> 5;
    const int mw = wave >> 1, nw = wave & 1;
    const int mbase = blockIdx.x * 128;
    const int nbase = blockIdx.y * 128;

    v8f acc[2][4] = {};

    auto load_tile = [&](int kt, int buf) {
        const int k0 = kt * 32;
#pragma unroll
        for (int j = 0; j < 2; ++j) {
            int i = tid + 256 * j;                 // 512 uint4 for 128x32 bf16 A tile
            int row = i >> 2, kq = i & 3;
            uint4 u = *(const uint4*)(A + (size_t)(mbase + row) * 1024 + k0 + kq * 8);
            *(uint4*)&As[buf][row][kq * 8] = u;
        }
#pragma unroll
        for (int j = 0; j < 4; ++j) {
            int i = tid + 256 * j;                 // 32(k) x 128(n), store transposed
            int kr = i >> 5, nq = i & 31;
            float4 f = *(const float4*)(w + (size_t)(k0 + kr) * 1024 + nbase + nq * 4);
            Bs[buf][nq * 4 + 0][kr] = (bf16)f.x;
            Bs[buf][nq * 4 + 1][kr] = (bf16)f.y;
            Bs[buf][nq * 4 + 2][kr] = (bf16)f.z;
            Bs[buf][nq * 4 + 3][kr] = (bf16)f.w;
        }
    };

    auto prefetch_tile = [&](int kt) {
        const int k0 = kt * 32;
        {
            int row = tid >> 1, kq = (tid & 1) * 16;
            __builtin_prefetch(A + (size_t)(mbase + row) * 1024 + k0 + kq, 0, 0);
        }
        {
            int kr = tid >> 3, nq = (tid & 7) * 16;
            __builtin_prefetch(w + (size_t)(k0 + kr) * 1024 + nbase + nq, 0, 0);
        }
    };

    prefetch_tile(1);
    load_tile(0, 0);
    __syncthreads();
    int buf = 0;
    for (int kt = 0; kt < 32; ++kt) {
        if (kt + 2 < 32) prefetch_tile(kt + 2);
        if (kt + 1 < 32) load_tile(kt + 1, buf ^ 1);
        v16bf a0 = load_frag_a(&As[buf][mw * 32][0], 40, lane);
        v16bf a1 = load_frag_a(&As[buf][mw * 32 + 16][0], 40, lane);
        v16bf b0 = load_frag_b(&Bs[buf][nw * 64 + 0][0], 40, lane);
        v16bf b1 = load_frag_b(&Bs[buf][nw * 64 + 16][0], 40, lane);
        v16bf b2 = load_frag_b(&Bs[buf][nw * 64 + 32][0], 40, lane);
        v16bf b3 = load_frag_b(&Bs[buf][nw * 64 + 48][0], 40, lane);
        acc[0][0] = WMMA_BF16(a0, b0, acc[0][0]);
        acc[1][0] = WMMA_BF16(a1, b0, acc[1][0]);
        acc[0][1] = WMMA_BF16(a0, b1, acc[0][1]);
        acc[1][1] = WMMA_BF16(a1, b1, acc[1][1]);
        acc[0][2] = WMMA_BF16(a0, b2, acc[0][2]);
        acc[1][2] = WMMA_BF16(a1, b2, acc[1][2]);
        acc[0][3] = WMMA_BF16(a0, b3, acc[0][3]);
        acc[1][3] = WMMA_BF16(a1, b3, acc[1][3]);
        buf ^= 1;
        __syncthreads();
    }

    const int hh = lane >> 4, nL = lane & 15;
#pragma unroll
    for (int fn = 0; fn < 4; ++fn) {
        const int col = nbase + nw * 64 + fn * 16 + nL;
        const float bv = bias[col];
#pragma unroll
        for (int fm = 0; fm < 2; ++fm) {
#pragma unroll
            for (int r = 0; r < 8; ++r) {
                const int row = mbase + mw * 32 + fm * 16 + r + 8 * hh;
                out[(size_t)row * 1024 + col] = acc[fm][fn][r] + bv;
            }
        }
    }
}

// ---------------------------------------------------------------------------
extern "C" void kernel_launch(void* const* d_in, const int* in_sizes, int n_in,
                              void* d_out, int out_size, void* d_ws, size_t ws_size,
                              hipStream_t stream) {
    const float* x     = (const float*)d_in[0];   // [8,1024,1024]
    const float* w_qkv = (const float*)d_in[1];   // [1024,3072]
    const float* w_out = (const float*)d_in[2];   // [1024,1024]
    const float* b_out = (const float*)d_in[3];   // [1024]
    float* out = (float*)d_out;                   // [8,1024,1024]

    char* ws = (char*)d_ws;
    const size_t SEG = (size_t)16 << 20;          // 16 MiB per buffer
    bf16* Q  = (bf16*)(ws);                       // [b*h][n][64], pre-scaled
    bf16* K  = (bf16*)(ws + SEG);                 // [b*h][n][64]
    bf16* Vt = (bf16*)(ws + 2 * SEG);             // [b*h][64][n]  (transposed)
    bf16* AO = (bf16*)(ws + 3 * SEG);             // [b][n][1024]

    qkv_gemm_kernel<<<dim3(64, 24), 256, 0, stream>>>(x, w_qkv, Q, K, Vt);
    attn_kernel<<<dim3(8 * 16 * 64), 32, 0, stream>>>(Q, K, Vt, AO);
    out_gemm_kernel<<<dim3(64, 8), 256, 0, stream>>>(AO, w_out, b_out, out);
}
